// MultiColorAlignmentBlock_84825604096510
// MI455X (gfx1250) — compile-verified, matrix operand order checked by reading, not run on hardware
//
#include <hip/hip_runtime.h>
#include <cstdint>
#include <cstddef>

// ---------------------------------------------------------------------------
// Problem constants (from reference)
// ---------------------------------------------------------------------------
constexpr int kH  = 192;
constexpr int kW  = 192;
constexpr int kHW = kH * kW;          // 36864
constexpr int kBlocks = 6;
constexpr int kWsize  = 15;           // channels per split block
constexpr int kStride = 5;            // (40-15)/(6-1)
constexpr int kNF0 = 32;
constexpr int kNF1 = 64;
constexpr int kOffC = 72;             // 18*GROUPS

typedef __attribute__((ext_vector_type(16))) _Float16 v16h;
typedef __attribute__((ext_vector_type(8)))  _Float16 v8h;
typedef __attribute__((ext_vector_type(8)))  float    v8f;

// (lane,half) -> K index inside a 16x32 f16 WMMA fragment (same mapping for A and B).
// lanes 0-15: V0..3 hold K=0..7, V4..7 hold K=16..23
// lanes 16-31: V0..3 hold K=8..15, V4..7 hold K=24..31
__device__ __forceinline__ int frag_k(int hiK, int h) {
    return (h < 8 ? h : h + 8) + hiK;   // hiK = 0 or 8
}

// ---------------------------------------------------------------------------
// Weight packing: f32 OIHW -> f16 WMMA A-fragments.
// Fragment f = (octile*9 + p)*kchunks + ch ; 512 f16 each, stored lane*16+h.
// Element (lane,h): M = lane&15 (oc), K = frag_k (ci within 32-chunk).
// ---------------------------------------------------------------------------
__global__ void pack_conv_w(const float* __restrict__ w, _Float16* __restrict__ wp,
                            int OC, int CIN, int kchunks) {
    int f = blockIdx.x;
    int ch = f % kchunks;
    int p  = (f / kchunks) % 9;
    int octile = f / (kchunks * 9);
    int t = threadIdx.x;              // 0..511
    int lane = t >> 4, h = t & 15;
    int m  = lane & 15;
    int hiK = (lane & 16) ? 8 : 0;
    int k  = frag_k(hiK, h);
    int oc = octile * 16 + m;
    int ci = ch * 32 + k;
    float v = 0.f;
    if (oc < OC && ci < CIN) v = w[((size_t)oc * CIN + ci) * 9 + p];
    wp[(size_t)f * 512 + t] = (_Float16)v;
}

// dcn weights: f32 [64][16][3][3] -> f16 [g][k][o][i] (i contiguous)
__global__ void pack_dcn_w(const float* __restrict__ w, _Float16* __restrict__ o16) {
    int idx = blockIdx.x * blockDim.x + threadIdx.x;   // 4*9*16*16 = 9216
    if (idx >= 4 * 9 * 16 * 16) return;
    int i = idx & 15;
    int o = (idx >> 4) & 15;
    int k = (idx >> 8) % 9;
    int g = idx / (9 * 256);
    o16[idx] = (_Float16)w[(((size_t)(g * 16 + o)) * 16 + i) * 9 + k];
}

// ---------------------------------------------------------------------------
// split_blocks + f32->f16 repack: x f32 [40,H,W] channels [c0,c0+15) -> [HW,32] f16
// ---------------------------------------------------------------------------
__global__ void split_pack(const float* __restrict__ x, _Float16* __restrict__ out, int c0) {
    int pix = blockIdx.x * blockDim.x + threadIdx.x;
    if (pix >= kHW) return;
    _Float16* o = out + (size_t)pix * 32;
    #pragma unroll
    for (int c = 0; c < 32; ++c)
        o[c] = (c < kWsize) ? (_Float16)x[(size_t)(c0 + c) * kHW + pix] : (_Float16)0.f;
}

// offset-conv input: [HW,160] f16 = [mask | other(64) | mid(64) | zeros(31)]
__global__ void offin_pack(const float* __restrict__ m0, const float* __restrict__ m1,
                           const _Float16* __restrict__ of, const _Float16* __restrict__ mf,
                           _Float16* __restrict__ out) {
    int pix = blockIdx.x * blockDim.x + threadIdx.x;
    if (pix >= kHW) return;
    float mv = m0[pix] + (m1 ? m1[pix] : 0.f);
    _Float16* o = out + (size_t)pix * 160;
    o[0] = (_Float16)mv;
    const _Float16* a = of + (size_t)pix * 64;
    const _Float16* b = mf + (size_t)pix * 64;
    #pragma unroll
    for (int c = 0; c < 64; ++c) o[1 + c]  = a[c];
    #pragma unroll
    for (int c = 0; c < 64; ++c) o[65 + c] = b[c];
    #pragma unroll
    for (int c = 129; c < 160; ++c) o[c] = (_Float16)0.f;
}

// zero pad channels 216..223 of fusion input [HW,224]
__global__ void fuse_padfill(_Float16* __restrict__ fusein) {
    int pix = blockIdx.x * blockDim.x + threadIdx.x;
    if (pix >= kHW) return;
    _Float16* o = fusein + (size_t)pix * 224 + 216;
    #pragma unroll
    for (int c = 0; c < 8; ++c) o[c] = (_Float16)0.f;
}

// mid feature copy: f16 [HW,64] -> f32 planar [64,HW] in d_out
__global__ void midcopy(const _Float16* __restrict__ feat, float* __restrict__ out) {
    int pix = blockIdx.x * blockDim.x + threadIdx.x;
    if (pix >= kHW) return;
    const _Float16* f = feat + (size_t)pix * 64;
    #pragma unroll
    for (int c = 0; c < 64; ++c) out[(size_t)c * kHW + pix] = (float)f[c];
}

// ---------------------------------------------------------------------------
// 3x3 SAME conv + LeakyReLU via WMMA implicit GEMM.
// Input:  X f16 [HW, Cp] (Cp multiple of 32, zero-padded channels).
// Block: 128 threads (4 waves). Block tile: 64 pixels (one row strip) x ALL
// output channels (NOCT tiles of 16, register accumulators). The 3x66x64
// input patch is staged once per 64-channel K-group via async LDS loads and
// reused by every OC tile. Per (p,chunk) step: preload all NOCT A fragments,
// then issue NOCT back-to-back WMMAs against the shared B fragment.
// ---------------------------------------------------------------------------
template<int NOCT>
__global__ __launch_bounds__(128)
void conv3x3_wmma(const _Float16* __restrict__ X, int Cp, int kchunks,
                  const _Float16* __restrict__ Wp, const float* __restrict__ bias,
                  int OC,
                  _Float16* __restrict__ outH, int OCp, int oc_off,
                  float* __restrict__ outF,
                  float slope) {
    __shared__ _Float16 lds[3 * 66 * 64];   // patch rows y-1..y+1, cols x0-1..x0+64, 64 ch
                                            // sole shared object -> LDS offset 0

    const int strip = blockIdx.x;           // 576 strips
    const int y  = strip / 3;
    const int x0 = (strip % 3) * 64;

    const int lane = threadIdx.x & 31;
    const int wv   = threadIdx.x >> 5;
    const int n    = lane & 15;
    const int hiK  = (lane & 16) ? 8 : 0;

    const size_t otStride = (size_t)9 * kchunks * 512;   // A-fragment stride per OC tile
    const _Float16* wlane = Wp + (size_t)lane * 16;

    v8f acc[NOCT];
    #pragma unroll
    for (int ot = 0; ot < NOCT; ++ot) acc[ot] = (v8f){};

    for (int c0 = 0; c0 < Cp; c0 += 64) {
        const int stC = (Cp - c0) < 64 ? (Cp - c0) : 64;   // 32 or 64
        __syncthreads();   // previous group fully consumed before overwrite

        // ---- stage channels [c0, c0+stC) of the 3x66 patch ----
        // interior: async global->LDS b128 (ASYNCcnt); border: zero ds_store.
        const int perSeg = stC >> 3;               // 16B transfers per segment
        const int total  = 3 * 66 * perSeg;
        for (int t = threadIdx.x; t < total; t += 128) {
            int seg = t / perSeg;
            int i   = t - seg * perSeg;
            int r = seg / 66, c = seg % 66;
            int gy = y + r - 1, gx = x0 + c - 1;
            if (gy >= 0 && gy < kH && gx >= 0 && gx < kW) {
                const _Float16* src = X + ((size_t)gy * kW + gx) * Cp + c0 + i * 8;
                unsigned ldsOff = (unsigned)(seg * 128 + i * 16);   // byte offset
                asm volatile("global_load_async_to_lds_b128 %0, %1, off"
                             :: "v"(ldsOff), "v"(src) : "memory");
            } else {
                v8h z = {};
                *(v8h*)(lds + seg * 64 + i * 8) = z;
            }
        }
        asm volatile("s_wait_asynccnt 0x0" ::: "memory");
        __syncthreads();

        // ---- compute ----
        const int chBase = c0 >> 5;
        const int nchl   = stC >> 5;
        #pragma unroll
        for (int p = 0; p < 9; ++p) {
            const int dy  = p / 3;
            const int dxk = p % 3;
            const _Float16* bbase = lds + ((dy * 66) + (wv * 16 + n + dxk)) * 64;
            for (int chl = 0; chl < nchl; ++chl) {
                const int gch = chBase + chl;
                // A fragments: NOCT independent contiguous 32B reads
                const _Float16* wbase = wlane + (size_t)(p * kchunks + gch) * 512;
                v16h af[NOCT];
                #pragma unroll
                for (int ot = 0; ot < NOCT; ++ot)
                    af[ot] = *(const v16h*)(wbase + (size_t)ot * otStride);
                // B fragment: two contiguous 16B LDS reads (shared by all OC tiles)
                const _Float16* bp = bbase + chl * 32 + hiK;
                v8h blo = *(const v8h*)(bp);
                v8h bhi = *(const v8h*)(bp + 16);
                v16h bf;
                #pragma unroll
                for (int i = 0; i < 8; ++i) { bf[i] = blo[i]; bf[i + 8] = bhi[i]; }
                // WMMA burst: NOCT independent accumulator chains
                #pragma unroll
                for (int ot = 0; ot < NOCT; ++ot)
                    acc[ot] = __builtin_amdgcn_wmma_f32_16x16x32_f16(
                                  false, af[ot], false, bf, (short)0, acc[ot], false, false);
            }
        }
    }

    // ---- epilogue: lane holds pixel n, OCs (ot*16 + hiK + v) ----
    const int px  = x0 + wv * 16 + n;
    const size_t pix = (size_t)y * kW + px;
    #pragma unroll
    for (int ot = 0; ot < NOCT; ++ot) {
        const int oc0 = ot * 16 + hiK;
        if (outH) {
            if (oc0 + 8 <= OC) {
                v8h o;
                #pragma unroll
                for (int v = 0; v < 8; ++v) {
                    float rr = acc[ot][v] + bias[oc0 + v];
                    rr = rr > 0.f ? rr : rr * slope;
                    o[v] = (_Float16)rr;
                }
                *(v8h*)(outH + pix * OCp + oc_off + oc0) = o;
            } else {
                #pragma unroll
                for (int v = 0; v < 8; ++v) {
                    int oc = oc0 + v;
                    if (oc < OC) {
                        float rr = acc[ot][v] + bias[oc];
                        rr = rr > 0.f ? rr : rr * slope;
                        outH[pix * OCp + oc_off + oc] = (_Float16)rr;
                    }
                }
            }
        } else {
            #pragma unroll
            for (int v = 0; v < 8; ++v) {
                int oc = oc0 + v;
                if (oc < OC) {
                    float rr = acc[ot][v] + bias[oc];
                    rr = rr > 0.f ? rr : rr * slope;
                    outF[(size_t)oc * kHW + pix] = rr;
                }
            }
        }
    }
}

// ---------------------------------------------------------------------------
// Grouped deformable conv (groups=4, k=9, pad=1). feat f16 [HW,64],
// offs f32 [72,HW] layout [g][k][{dy,dx}], wgt f16 [g][k][o][i], out f32 [64,HW].
// ---------------------------------------------------------------------------
__global__ __launch_bounds__(128)
void deform_conv(const _Float16* __restrict__ feat,
                 const float* __restrict__ offs,
                 const _Float16* __restrict__ wgt,
                 const float* __restrict__ bias,
                 float* __restrict__ out) {
    __shared__ _Float16 wl[4 * 9 * 16 * 16];   // 9216 f16 = 18KB
    for (int i = threadIdx.x; i < 4 * 9 * 16 * 16; i += blockDim.x) wl[i] = wgt[i];
    __syncthreads();

    int pix = blockIdx.x * blockDim.x + threadIdx.x;
    if (pix >= kHW) return;
    int y = pix / kW, x = pix % kW;

    float acc[64];
    #pragma unroll
    for (int i = 0; i < 64; ++i) acc[i] = 0.f;

    for (int g = 0; g < 4; ++g) {
        for (int k = 0; k < 9; ++k) {
            float dy = offs[(size_t)((g * 9 + k) * 2 + 0) * kHW + pix];
            float dx = offs[(size_t)((g * 9 + k) * 2 + 1) * kHW + pix];
            float py = (float)y + (float)(k / 3 - 1) + dy;
            float px = (float)x + (float)(k % 3 - 1) + dx;
            float y0f = floorf(py), x0f = floorf(px);
            float ly = py - y0f, lx = px - x0f;
            int y0 = (int)y0f, xx0 = (int)x0f;

            float smp[16];
            #pragma unroll
            for (int i = 0; i < 16; ++i) smp[i] = 0.f;

            #pragma unroll
            for (int cc = 0; cc < 4; ++cc) {
                int ddy = cc >> 1, ddx = cc & 1;
                int yc = y0 + ddy, xc = xx0 + ddx;
                float w2 = (ddy ? ly : 1.f - ly) * (ddx ? lx : 1.f - lx);
                if (yc >= 0 && yc < kH && xc >= 0 && xc < kW && w2 != 0.f) {
                    const _Float16* fp = feat + ((size_t)yc * kW + xc) * 64 + g * 16;
                    v8h f0 = *(const v8h*)(fp);
                    v8h f1 = *(const v8h*)(fp + 8);
                    #pragma unroll
                    for (int i = 0; i < 8; ++i) {
                        smp[i]     += w2 * (float)f0[i];
                        smp[i + 8] += w2 * (float)f1[i];
                    }
                }
            }

            const _Float16* wk = wl + ((g * 9 + k) * 16) * 16;
            #pragma unroll
            for (int o = 0; o < 16; ++o) {
                const v8h w0 = *(const v8h*)(wk + o * 16);
                const v8h w1 = *(const v8h*)(wk + o * 16 + 8);
                float s = 0.f;
                #pragma unroll
                for (int i = 0; i < 8; ++i) {
                    s += smp[i] * (float)w0[i];
                    s += smp[i + 8] * (float)w1[i];
                }
                acc[g * 16 + o] += s;
            }
        }
    }
    #pragma unroll
    for (int c = 0; c < 64; ++c) out[(size_t)c * kHW + pix] = acc[c] + bias[c];
}

// ---------------------------------------------------------------------------
// Host orchestration
// ---------------------------------------------------------------------------
extern "C" void kernel_launch(void* const* d_in, const int* in_sizes, int n_in,
                              void* d_out, int out_size, void* d_ws, size_t ws_size,
                              hipStream_t stream) {
    (void)in_sizes; (void)n_in; (void)out_size; (void)ws_size;

    // --- inputs (setup_inputs insertion order, recursively flattened) ---
    const float* rgbin[3] = { (const float*)d_in[0], (const float*)d_in[1], (const float*)d_in[2] };
    const float* mask = (const float*)d_in[3];
    auto Wd = [&](int c, int l) { return (const float*)d_in[4 + c * 8 + l * 2]; };
    auto Bd = [&](int c, int l) { return (const float*)d_in[5 + c * 8 + l * 2]; };
    const float* Woff[3] = { (const float*)d_in[28], (const float*)d_in[30], (const float*)d_in[32] };
    const float* Boff[3] = { (const float*)d_in[29], (const float*)d_in[31], (const float*)d_in[33] };
    const float* Wfuse = (const float*)d_in[34];
    const float* Bfuse = (const float*)d_in[35];
    const float* Wdcn[3] = { (const float*)d_in[36], (const float*)d_in[38], (const float*)d_in[40] };
    const float* Bdcn[3] = { (const float*)d_in[37], (const float*)d_in[39], (const float*)d_in[41] };

    float* out = (float*)d_out;

    // --- workspace carve ---
    char* ws = (char*)d_ws;
    size_t off = 0;
    auto carve = [&](size_t bytes) -> void* {
        off = (off + 255) & ~(size_t)255;
        void* p = ws + off;
        off += bytes;
        return p;
    };
    _Float16* feat[3][6];
    for (int c = 0; c < 3; ++c)
        for (int b = 0; b < 6; ++b)
            feat[c][b] = (_Float16*)carve((size_t)kHW * 64 * 2);
    _Float16* tmpS = (_Float16*)carve((size_t)kHW * 32 * 2);
    _Float16* tmp0 = (_Float16*)carve((size_t)kHW * 32 * 2);
    _Float16* tmp1 = (_Float16*)carve((size_t)kHW * 32 * 2);
    _Float16* offin[3];
    for (int c = 0; c < 3; ++c) offin[c] = (_Float16*)carve((size_t)kHW * 160 * 2);
    _Float16* fusein = (_Float16*)carve((size_t)kHW * 224 * 2);
    float*    offout = (float*)carve((size_t)kOffC * kHW * 4);
    _Float16* wpDown[3][4];
    const int dOct[4] = { 2, 2, 2, 4 };   // OC tiles per down layer (32,32,32,64)
    for (int c = 0; c < 3; ++c)
        for (int l = 0; l < 4; ++l)
            wpDown[c][l] = (_Float16*)carve((size_t)dOct[l] * 9 * 1 * 512 * 2);
    _Float16* wpOff[3];
    for (int c = 0; c < 3; ++c) wpOff[c] = (_Float16*)carve((size_t)5 * 9 * 5 * 512 * 2);
    _Float16* wpFuse = (_Float16*)carve((size_t)5 * 9 * 7 * 512 * 2);
    _Float16* wpDcn[3];
    for (int c = 0; c < 3; ++c) wpDcn[c] = (_Float16*)carve((size_t)9216 * 2);

    // --- pack weights (deterministic, every call) ---
    auto packw = [&](const float* w, _Float16* wp, int OC, int CIN, int octiles, int kch) {
        pack_conv_w<<<octiles * 9 * kch, 512, 0, stream>>>(w, wp, OC, CIN, kch);
    };
    const int dCIN[4] = { kWsize, kNF0, kNF0, kNF0 };
    const int dOC [4] = { kNF0, kNF0, kNF0, kNF1 };
    for (int c = 0; c < 3; ++c)
        for (int l = 0; l < 4; ++l)
            packw(Wd(c, l), wpDown[c][l], dOC[l], dCIN[l], dOct[l], 1);
    for (int c = 0; c < 3; ++c) packw(Woff[c], wpOff[c], kOffC, 129, 5, 5);
    packw(Wfuse, wpFuse, kOffC, 216, 5, 7);
    for (int c = 0; c < 3; ++c)
        pack_dcn_w<<<(9216 + 255) / 256, 256, 0, stream>>>(Wdcn[c], wpDcn[c]);

    auto conv = [&](const _Float16* X, int Cp, int kch, const _Float16* Wp, const float* bias,
                    int OC, int octiles, _Float16* oh, int OCp, int ocoff, float* of, float slope) {
        dim3 grid(576, 1, 1);
        if (octiles == 2)
            conv3x3_wmma<2><<<grid, 128, 0, stream>>>(X, Cp, kch, Wp, bias, OC, oh, OCp, ocoff, of, slope);
        else if (octiles == 4)
            conv3x3_wmma<4><<<grid, 128, 0, stream>>>(X, Cp, kch, Wp, bias, OC, oh, OCp, ocoff, of, slope);
        else
            conv3x3_wmma<5><<<grid, 128, 0, stream>>>(X, Cp, kch, Wp, bias, OC, oh, OCp, ocoff, of, slope);
    };

    // --- down path: 3 colors x 6 blocks x 4 layers ---
    for (int c = 0; c < 3; ++c) {
        for (int b = 0; b < kBlocks; ++b) {
            split_pack<<<kHW / 256, 256, 0, stream>>>(rgbin[c], tmpS, b * kStride);
            conv(tmpS, 32, 1, wpDown[c][0], Bd(c, 0), 32, 2, tmp0, 32, 0, nullptr, 0.2f);
            conv(tmp0, 32, 1, wpDown[c][1], Bd(c, 1), 32, 2, tmp1, 32, 0, nullptr, 0.2f);
            conv(tmp1, 32, 1, wpDown[c][2], Bd(c, 2), 32, 2, tmp0, 32, 0, nullptr, 0.2f);
            conv(tmp0, 32, 1, wpDown[c][3], Bd(c, 3), 64, 4, feat[c][b], 64, 0, nullptr, 0.2f);
        }
    }

    // --- mid features -> output channel group 3 ---
    const int mid = kBlocks / 2;   // 3
    for (int c = 0; c < 3; ++c)
        midcopy<<<kHW / 256, 256, 0, stream>>>(feat[c][mid],
                                               out + (size_t)c * 384 * kHW + (size_t)mid * 64 * kHW);

    // --- fusion-input pad channels (216..223) zero, once ---
    fuse_padfill<<<kHW / 256, 256, 0, stream>>>(fusein);

    // --- per non-mid block: offsets -> fusion -> 3 deformable convs ---
    for (int j = 0; j < kBlocks - 1; ++j) {
        const int bidx = (j < mid) ? j : j + 1;   // source block (skip mid)
        const int grp  = bidx;                    // output channel group (same skip)

        offin_pack<<<kHW / 256, 256, 0, stream>>>(mask + 0 * kHW, (const float*)nullptr,
                                                  feat[0][bidx], feat[0][mid], offin[0]);
        offin_pack<<<kHW / 256, 256, 0, stream>>>(mask + 1 * kHW, mask + 2 * kHW,
                                                  feat[1][bidx], feat[1][mid], offin[1]);
        offin_pack<<<kHW / 256, 256, 0, stream>>>(mask + 3 * kHW, (const float*)nullptr,
                                                  feat[2][bidx], feat[2][mid], offin[2]);

        conv(offin[0], 160, 5, wpOff[0], Boff[0], kOffC, 5, fusein, 224, 0,   nullptr, 0.1f);
        conv(offin[1], 160, 5, wpOff[1], Boff[1], kOffC, 5, fusein, 224, 72,  nullptr, 0.1f);
        conv(offin[2], 160, 5, wpOff[2], Boff[2], kOffC, 5, fusein, 224, 144, nullptr, 0.1f);

        conv(fusein, 224, 7, wpFuse, Bfuse, kOffC, 5, nullptr, 0, 0, offout, 0.2f);

        for (int c = 0; c < 3; ++c) {
            float* dst = out + (size_t)c * 384 * kHW + (size_t)grp * 64 * kHW;
            deform_conv<<<kHW / 128, 128, 0, stream>>>(feat[c][bidx], offout, wpDcn[c],
                                                       Bdcn[c], dst);
        }
    }
}
